// MoELayer_87179246175009
// MI455X (gfx1250) — matrix operand order repaired and checked by reading, not compiled, not run on hardware
//
#include <hip/hip_runtime.h>

// ---------------------------------------------------------------------------
// MoE layer (LN -> router -> top-2 routed expert FFN -> residual) for gfx1250.
// f16 WMMA (v_wmma_f32_16x16x32_f16) with f32 accumulation; token gather per
// expert; weights pre-converted/transposed to f16 so all LDS staging is
// contiguous 16B copies via GLOBAL_LOAD_ASYNC_TO_LDS_B128 (ASYNCcnt path).
// Gather/source addresses are hoisted out of the K loop.
// ---------------------------------------------------------------------------

#define D_MODEL   1024
#define D_FF      4096
#define N_EXP     8
#define N_TOK     8192            // B*T = 4*2048
#define N_SLOTS   (2 * N_TOK)     // top-2 routing slots
#define LN_EPS    1e-5f

#define TILE_M    128
#define TILE_N    128
#define KSTEP     32
#define MAX_TILES (N_TOK / TILE_M)   // 64 tiles worst case per expert
#define AB_STRIDE 40                 // halfs per LDS row: 32 data + 8 pad (80B = 20 banks)
#define BUF_HALF  10240              // bytes of one A (or B) stage buffer
#define BUF_PAIR  20480              // bytes of one A+B stage buffer pair
#define SMEM_BYTES 67584             // 128*132*4 f32 out tile (largest user)

typedef __attribute__((ext_vector_type(16))) _Float16 v16h;
typedef __attribute__((ext_vector_type(8)))  _Float16 v8h;
typedef __attribute__((ext_vector_type(8)))  float    v8f;
typedef int vs4i __attribute__((vector_size(16)));   // builtin's b128 payload type

// ---------------------------------------------------------------------------
// Async global->LDS copy (CDNA5 ASYNCcnt path) with synchronous fallback.
// ---------------------------------------------------------------------------
#if __has_builtin(__builtin_amdgcn_global_load_async_to_lds_b128)
#define HAS_ASYNC_LDS 1
#else
#define HAS_ASYNC_LDS 0
#endif

__device__ __forceinline__ void cp16_g2l(const void* g, void* l) {
#if HAS_ASYNC_LDS
  __builtin_amdgcn_global_load_async_to_lds_b128(
      (__attribute__((address_space(1))) vs4i*)g,
      (__attribute__((address_space(3))) vs4i*)l, 0, 0);
#else
  *(uint4*)l = *(const uint4*)g;
#endif
}

__device__ __forceinline__ void cp_wait() {
#if HAS_ASYNC_LDS
#if __has_builtin(__builtin_amdgcn_s_wait_asynccnt)
  __builtin_amdgcn_s_wait_asynccnt(0);
#else
  asm volatile("s_wait_asynccnt 0x0" ::: "memory");
#endif
#endif
}

// ---------------------------------------------------------------------------
// WMMA helpers
// ---------------------------------------------------------------------------
__device__ __forceinline__ v8f wmma_f16(v16h a, v16h b, v8f c) {
  // D = A(16x32 f16) * B(32x16 f16) + C(16x16 f32)
  return __builtin_amdgcn_wmma_f32_16x16x32_f16(
      /*neg_a=*/false, a, /*neg_b=*/false, b,
      /*c_mod=*/(short)0, c, /*reuse_a=*/false, /*reuse_b=*/false);
}

// A fragment (16x32, row-major in LDS, row stride AB_STRIDE halfs).
// lane<16: M=lane,     K = {0..7, 16..23}
// lane>=16: M=lane-16, K = {8..15, 24..31}
__device__ __forceinline__ v16h frag_a(const _Float16* buf, int row_base, int lane) {
  const int hs = lane >> 4;
  const _Float16* p = buf + (size_t)(row_base + (lane & 15)) * AB_STRIDE + hs * 8;
  v8h lo = *(const v8h*)(p);        // K = hs*8 + 0..7
  v8h hi = *(const v8h*)(p + 16);   // K = hs*8 + 16..23
  return __builtin_shufflevector(lo, hi, 0, 1, 2, 3, 4, 5, 6, 7,
                                 8, 9, 10, 11, 12, 13, 14, 15);
}

// B fragment: LDS holds B transposed as [n][k] (row stride AB_STRIDE halfs).
// lane<16: N=lane, K = {0..7, 16..23} ; lane>=16: N=lane-16, K = {8..15, 24..31}
__device__ __forceinline__ v16h frag_b(const _Float16* buf, int row_base, int lane) {
  const int kb = (lane >> 4) * 8;
  const _Float16* p = buf + (size_t)(row_base + (lane & 15)) * AB_STRIDE + kb;
  v8h lo = *(const v8h*)(p);
  v8h hi = *(const v8h*)(p + 16);
  return __builtin_shufflevector(lo, hi, 0, 1, 2, 3, 4, 5, 6, 7,
                                 8, 9, 10, 11, 12, 13, 14, 15);
}

// ---------------------------------------------------------------------------
// Weight prep: transpose + f32 -> f16.  src[R][C] f32 -> dst[C][R] f16.
// grid = (C/32, R/32, E), 256 threads.
// ---------------------------------------------------------------------------
__global__ __launch_bounds__(256)
void moe_prep_w(const float* __restrict__ src, _Float16* __restrict__ dst,
                int R, int C) {
  __shared__ _Float16 tile[32][36];
  const size_t esz = (size_t)R * C;
  const float* s = src + (size_t)blockIdx.z * esz;
  _Float16* d = dst + (size_t)blockIdx.z * esz;
  const int c0 = blockIdx.x * 32, r0 = blockIdx.y * 32;
  const int r  = threadIdx.x >> 3;          // 0..31
  const int c4 = (threadIdx.x & 7) * 4;     // 0,4,...,28

  const float4 v = *(const float4*)(s + (size_t)(r0 + r) * C + c0 + c4);
  tile[r][c4 + 0] = (_Float16)v.x;
  tile[r][c4 + 1] = (_Float16)v.y;
  tile[r][c4 + 2] = (_Float16)v.z;
  tile[r][c4 + 3] = (_Float16)v.w;
  __syncthreads();

  union { _Float16 h[4]; uint2 u; } o;      // dst[c0+r][r0+c4 .. +3]
  o.h[0] = tile[c4 + 0][r];
  o.h[1] = tile[c4 + 1][r];
  o.h[2] = tile[c4 + 2][r];
  o.h[3] = tile[c4 + 3][r];
  *(uint2*)(d + (size_t)(c0 + r) * R + r0 + c4) = o.u;
}

// ---------------------------------------------------------------------------
// Kernel 1: LayerNorm + router (softmax, top-2), writes xn as f16.
// One 256-thread block per token.
// ---------------------------------------------------------------------------
__global__ __launch_bounds__(256)
void moe_ln_router(const float* __restrict__ x, const float* __restrict__ gamma,
                   const float* __restrict__ beta, const float* __restrict__ rw,
                   const float* __restrict__ rb, _Float16* __restrict__ xn,
                   int* __restrict__ counts, int* __restrict__ topi,
                   float* __restrict__ topw, float* __restrict__ probs) {
  const int n = blockIdx.x;
  const int tid = threadIdx.x;
  __shared__ float2 red[256];
  __shared__ float  sbuf[256];
  __shared__ float  logits[N_EXP];

  const float4 xv = *(const float4*)(x + (size_t)n * D_MODEL + tid * 4);
  float xa[4] = {xv.x, xv.y, xv.z, xv.w};
  float s = xa[0] + xa[1] + xa[2] + xa[3];
  float ss = xa[0] * xa[0] + xa[1] * xa[1] + xa[2] * xa[2] + xa[3] * xa[3];
  red[tid] = make_float2(s, ss);
  __syncthreads();
  for (int o = 128; o > 0; o >>= 1) {
    if (tid < o) { red[tid].x += red[tid + o].x; red[tid].y += red[tid + o].y; }
    __syncthreads();
  }
  const float mu   = red[0].x * (1.0f / D_MODEL);
  const float var  = red[0].y * (1.0f / D_MODEL) - mu * mu;
  const float rstd = rsqrtf(var + LN_EPS);

  const float4 gv = *(const float4*)(gamma + tid * 4);
  const float4 bv = *(const float4*)(beta + tid * 4);
  float ga[4] = {gv.x, gv.y, gv.z, gv.w};
  float ba[4] = {bv.x, bv.y, bv.z, bv.w};
  float xnv[4];
  union { _Float16 h[4]; uint2 u; } pk;
#pragma unroll
  for (int j = 0; j < 4; ++j) {
    xnv[j] = (xa[j] - mu) * rstd * ga[j] + ba[j];
    pk.h[j] = (_Float16)xnv[j];
  }
  *(uint2*)(xn + (size_t)n * D_MODEL + tid * 4) = pk.u;

  // Router partial dot products: logits[e] = sum_d xn[d] * rw[d][e]
  float le[N_EXP];
#pragma unroll
  for (int e = 0; e < N_EXP; ++e) le[e] = 0.0f;
#pragma unroll
  for (int j = 0; j < 4; ++j) {
    const float* wr = rw + (size_t)(tid * 4 + j) * N_EXP;
#pragma unroll
    for (int e = 0; e < N_EXP; ++e) le[e] += xnv[j] * wr[e];
  }
#pragma unroll
  for (int e = 0; e < N_EXP; ++e) {
    sbuf[tid] = le[e];
    __syncthreads();
    for (int o = 128; o > 0; o >>= 1) {
      if (tid < o) sbuf[tid] += sbuf[tid + o];
      __syncthreads();
    }
    if (tid == 0) logits[e] = sbuf[0] + rb[e];
    __syncthreads();
  }

  if (tid == 0) {
    float mx = logits[0];
#pragma unroll
    for (int e = 1; e < N_EXP; ++e) mx = fmaxf(mx, logits[e]);
    float p[N_EXP], sum = 0.0f;
#pragma unroll
    for (int e = 0; e < N_EXP; ++e) { p[e] = __expf(logits[e] - mx); sum += p[e]; }
    const float inv = 1.0f / sum;
#pragma unroll
    for (int e = 0; e < N_EXP; ++e) probs[(size_t)n * N_EXP + e] = p[e] * inv;
    int i0 = 0;
#pragma unroll
    for (int e = 1; e < N_EXP; ++e) if (p[e] > p[i0]) i0 = e;
    int i1 = (i0 == 0) ? 1 : 0;
#pragma unroll
    for (int e = 0; e < N_EXP; ++e) if (e != i0 && p[e] > p[i1]) i1 = e;
    const float wsum = p[i0] + p[i1];
    topi[2 * n]     = i0;
    topi[2 * n + 1] = i1;
    topw[2 * n]     = p[i0] / wsum;
    topw[2 * n + 1] = p[i1] / wsum;
    atomicAdd(&counts[i0], 1);
    atomicAdd(&counts[i1], 1);
  }
}

// ---------------------------------------------------------------------------
// Tiny routing bookkeeping kernels
// ---------------------------------------------------------------------------
__global__ void moe_init(int* __restrict__ counts) {
  if (threadIdx.x < N_EXP) counts[threadIdx.x] = 0;
}

__global__ void moe_offsets(const int* __restrict__ counts,
                            int* __restrict__ offsets, int* __restrict__ cursors) {
  if (threadIdx.x == 0) {
    int a = 0;
    for (int e = 0; e < N_EXP; ++e) { offsets[e] = a; cursors[e] = a; a += counts[e]; }
    offsets[N_EXP] = a;
  }
}

__global__ void moe_scatter(const int* __restrict__ topi, int* __restrict__ cursors,
                            int* __restrict__ tok_of_slot, int* __restrict__ tok2slot) {
  const int n = blockIdx.x * blockDim.x + threadIdx.x;
  if (n >= N_TOK) return;
#pragma unroll
  for (int j = 0; j < 2; ++j) {
    const int e = topi[2 * n + j];
    const int s = atomicAdd(&cursors[e], 1);
    tok_of_slot[s] = n;
    tok2slot[2 * n + j] = s;
  }
}

// ---------------------------------------------------------------------------
// GEMM1: H[slot, f] = silu( xn[token(slot)] @ w1[e] + b1[e] )   (f16 out)
// B = w1t (f16, [F][D] per expert). grid = (F/128, E*MAX_TILES), 256 threads.
// ---------------------------------------------------------------------------
__global__ __launch_bounds__(256)
void moe_gemm1(const _Float16* __restrict__ xn, const _Float16* __restrict__ w1t,
               const float* __restrict__ b1, _Float16* __restrict__ H,
               const int* __restrict__ counts, const int* __restrict__ offsets,
               const int* __restrict__ tok_of_slot) {
  const int e   = blockIdx.y / MAX_TILES;
  const int m0  = (blockIdx.y % MAX_TILES) * TILE_M;
  const int cnt = counts[e];
  if (m0 >= cnt) return;
  const int off = offsets[e];
  const int f0  = blockIdx.x * TILE_N;
  const _Float16* w1te = w1t + (size_t)e * D_MODEL * D_FF;

  __shared__ __align__(16) unsigned char smem[SMEM_BYTES];
  int* toks = (int*)(smem + 2 * BUF_PAIR);   // 512 B past the stage buffers

  if (threadIdx.x < TILE_M) {
    int r = m0 + (int)threadIdx.x;
    if (r >= cnt) r = cnt - 1;                    // clamp padding rows
    toks[threadIdx.x] = tok_of_slot[off + r];
  }
  __syncthreads();

  // Hoist per-thread staging addresses (loop-invariant; only k0 advances).
  const int id0 = threadIdx.x,        row0 = id0 >> 2, kc0 = (id0 & 3) * 8;
  const int id1 = threadIdx.x + 256,  row1 = id1 >> 2, kc1 = (id1 & 3) * 8;
  const _Float16* gA0 = xn + (size_t)toks[row0] * D_MODEL + kc0;
  const _Float16* gA1 = xn + (size_t)toks[row1] * D_MODEL + kc1;
  const _Float16* gB0 = w1te + (size_t)(f0 + row0) * D_MODEL + kc0;
  const _Float16* gB1 = w1te + (size_t)(f0 + row1) * D_MODEL + kc1;
  _Float16* lA0 = (_Float16*)smem + (size_t)row0 * AB_STRIDE + kc0;
  _Float16* lA1 = (_Float16*)smem + (size_t)row1 * AB_STRIDE + kc1;
  _Float16* lB0 = (_Float16*)(smem + BUF_HALF) + (size_t)row0 * AB_STRIDE + kc0;
  _Float16* lB1 = (_Float16*)(smem + BUF_HALF) + (size_t)row1 * AB_STRIDE + kc1;
  __syncthreads();   // toks no longer needed after address hoist

  auto stage = [&](int bi, int k0) {
    const size_t boff = (size_t)bi * BUF_PAIR;    // bytes
    cp16_g2l(gA0 + k0, (unsigned char*)lA0 + boff);
    cp16_g2l(gB0 + k0, (unsigned char*)lB0 + boff);
    cp16_g2l(gA1 + k0, (unsigned char*)lA1 + boff);
    cp16_g2l(gB1 + k0, (unsigned char*)lB1 + boff);
  };

  const int lane = threadIdx.x & 31;
  const int wave = threadIdx.x >> 5;
  const int wm = (wave & 1) * 64;                 // wave tile: 64M x 32N
  const int wn = (wave >> 1) * 32;

  v8f acc[4][2];
  const v8f zero = {0, 0, 0, 0, 0, 0, 0, 0};
#pragma unroll
  for (int i = 0; i < 4; ++i) { acc[i][0] = zero; acc[i][1] = zero; }

  stage(0, 0);
  cp_wait();
  __syncthreads();
  const int NK = D_MODEL / KSTEP;                 // 32
  for (int ks = 0; ks < NK; ++ks) {
    const int cur = ks & 1;
    if (ks + 1 < NK) stage(cur ^ 1, (ks + 1) * KSTEP);
    const _Float16* Ac = (const _Float16*)(smem + (size_t)cur * BUF_PAIR);
    const _Float16* Bc = (const _Float16*)(smem + (size_t)cur * BUF_PAIR + BUF_HALF);
    v16h bf0 = frag_b(Bc, wn, lane);
    v16h bf1 = frag_b(Bc, wn + 16, lane);
#pragma unroll
    for (int i = 0; i < 4; ++i) {
      v16h af = frag_a(Ac, wm + i * 16, lane);
      acc[i][0] = wmma_f16(af, bf0, acc[i][0]);
      acc[i][1] = wmma_f16(af, bf1, acc[i][1]);
    }
    cp_wait();
    __syncthreads();
  }

  // Epilogue: +b1, silu, stage f16 tile in LDS, coalesced b128 stores.
  _Float16* OT = (_Float16*)smem;                 // 128 x 136 halfs
#pragma unroll
  for (int j = 0; j < 2; ++j) {
    const int col = wn + j * 16 + (lane & 15);
    const float bias = b1[(size_t)e * D_FF + f0 + col];
#pragma unroll
    for (int i = 0; i < 4; ++i) {
      const int rbase = wm + i * 16 + (lane >> 4) * 8;
#pragma unroll
      for (int r = 0; r < 8; ++r) {
        float v = acc[i][j][r] + bias;
        v = v / (1.0f + __expf(-v));              // silu
        OT[(size_t)(rbase + r) * 136 + col] = (_Float16)v;
      }
    }
  }
  __syncthreads();
#pragma unroll
  for (int t = 0; t < 8; ++t) {
    const int id = threadIdx.x + t * 256;         // 0..2047
    const int row = id >> 4, col = (id & 15) * 8;
    if (m0 + row < cnt) {
      const size_t slot = (size_t)(off + m0 + row);
      *(uint4*)(H + slot * D_FF + f0 + col) = *(const uint4*)(OT + (size_t)row * 136 + col);
    }
  }
}

// ---------------------------------------------------------------------------
// GEMM2: Y[slot, d] = H[slot] @ w2[e] + b2[e]   (f32 out)
// B = w2t (f16, [D][F] per expert). grid = (D/128, E*MAX_TILES)
// ---------------------------------------------------------------------------
__global__ __launch_bounds__(256)
void moe_gemm2(const _Float16* __restrict__ H, const _Float16* __restrict__ w2t,
               const float* __restrict__ b2, float* __restrict__ Y,
               const int* __restrict__ counts, const int* __restrict__ offsets) {
  const int e   = blockIdx.y / MAX_TILES;
  const int m0  = (blockIdx.y % MAX_TILES) * TILE_M;
  const int cnt = counts[e];
  if (m0 >= cnt) return;
  const int off = offsets[e];
  const int d0  = blockIdx.x * TILE_N;
  const _Float16* w2te = w2t + (size_t)e * D_FF * D_MODEL;

  __shared__ __align__(16) unsigned char smem[SMEM_BYTES];

  // Hoist per-thread staging addresses (loop-invariant; only k0 advances).
  const int id0 = threadIdx.x,        row0 = id0 >> 2, kc0 = (id0 & 3) * 8;
  const int id1 = threadIdx.x + 256,  row1 = id1 >> 2, kc1 = (id1 & 3) * 8;
  int r0c = m0 + row0; if (r0c >= cnt) r0c = cnt - 1;
  int r1c = m0 + row1; if (r1c >= cnt) r1c = cnt - 1;
  const _Float16* gA0 = H + (size_t)(off + r0c) * D_FF + kc0;
  const _Float16* gA1 = H + (size_t)(off + r1c) * D_FF + kc1;
  const _Float16* gB0 = w2te + (size_t)(d0 + row0) * D_FF + kc0;
  const _Float16* gB1 = w2te + (size_t)(d0 + row1) * D_FF + kc1;
  _Float16* lA0 = (_Float16*)smem + (size_t)row0 * AB_STRIDE + kc0;
  _Float16* lA1 = (_Float16*)smem + (size_t)row1 * AB_STRIDE + kc1;
  _Float16* lB0 = (_Float16*)(smem + BUF_HALF) + (size_t)row0 * AB_STRIDE + kc0;
  _Float16* lB1 = (_Float16*)(smem + BUF_HALF) + (size_t)row1 * AB_STRIDE + kc1;

  auto stage = [&](int bi, int k0) {
    const size_t boff = (size_t)bi * BUF_PAIR;    // bytes
    cp16_g2l(gA0 + k0, (unsigned char*)lA0 + boff);
    cp16_g2l(gB0 + k0, (unsigned char*)lB0 + boff);
    cp16_g2l(gA1 + k0, (unsigned char*)lA1 + boff);
    cp16_g2l(gB1 + k0, (unsigned char*)lB1 + boff);
  };

  const int lane = threadIdx.x & 31;
  const int wave = threadIdx.x >> 5;
  const int wm = (wave & 1) * 64;
  const int wn = (wave >> 1) * 32;

  v8f acc[4][2];
  const v8f zero = {0, 0, 0, 0, 0, 0, 0, 0};
#pragma unroll
  for (int i = 0; i < 4; ++i) { acc[i][0] = zero; acc[i][1] = zero; }

  stage(0, 0);
  cp_wait();
  __syncthreads();
  const int NK = D_FF / KSTEP;                    // 128
  for (int ks = 0; ks < NK; ++ks) {
    const int cur = ks & 1;
    if (ks + 1 < NK) stage(cur ^ 1, (ks + 1) * KSTEP);
    const _Float16* Ac = (const _Float16*)(smem + (size_t)cur * BUF_PAIR);
    const _Float16* Bc = (const _Float16*)(smem + (size_t)cur * BUF_PAIR + BUF_HALF);
    v16h bf0 = frag_b(Bc, wn, lane);
    v16h bf1 = frag_b(Bc, wn + 16, lane);
#pragma unroll
    for (int i = 0; i < 4; ++i) {
      v16h af = frag_a(Ac, wm + i * 16, lane);
      acc[i][0] = wmma_f16(af, bf0, acc[i][0]);
      acc[i][1] = wmma_f16(af, bf1, acc[i][1]);
    }
    cp_wait();
    __syncthreads();
  }

  // Epilogue: +b2, stage f32 tile in LDS, coalesced float4 stores.
  float* OTf = (float*)smem;                      // 128 x 132 floats
#pragma unroll
  for (int j = 0; j < 2; ++j) {
    const int col = wn + j * 16 + (lane & 15);
    const float bias = b2[(size_t)e * D_MODEL + d0 + col];
#pragma unroll
    for (int i = 0; i < 4; ++i) {
      const int rbase = wm + i * 16 + (lane >> 4) * 8;
#pragma unroll
      for (int r = 0; r < 8; ++r)
        OTf[(size_t)(rbase + r) * 132 + col] = acc[i][j][r] + bias;
    }
  }
  __syncthreads();
#pragma unroll
  for (int t = 0; t < 16; ++t) {
    const int id = threadIdx.x + t * 256;         // 0..4095
    const int row = id >> 5, col = (id & 31) * 4;
    if (m0 + row < cnt) {
      const size_t slot = (size_t)(off + m0 + row);
      *(float4*)(Y + slot * D_MODEL + d0 + col) =
          *(const float4*)(OTf + (size_t)row * 132 + col);
    }
  }
}

// ---------------------------------------------------------------------------
// Combine: out = residual + w0*Y[s0] + w1*Y[s1]
// ---------------------------------------------------------------------------
__global__ __launch_bounds__(256)
void moe_combine(const float* __restrict__ x, const float* __restrict__ Y,
                 const int* __restrict__ tok2slot, const float* __restrict__ topw,
                 float* __restrict__ out) {
  const int n = blockIdx.x;
  const int c = threadIdx.x * 4;
  const int s0 = tok2slot[2 * n], s1 = tok2slot[2 * n + 1];
  const float w0 = topw[2 * n], w1 = topw[2 * n + 1];
  const float4 xv = *(const float4*)(x + (size_t)n * D_MODEL + c);
  const float4 y0 = *(const float4*)(Y + (size_t)s0 * D_MODEL + c);
  const float4 y1 = *(const float4*)(Y + (size_t)s1 * D_MODEL + c);
  float4 o;
  o.x = xv.x + w0 * y0.x + w1 * y1.x;
  o.y = xv.y + w0 * y0.y + w1 * y1.y;
  o.z = xv.z + w0 * y0.z + w1 * y1.z;
  o.w = xv.w + w0 * y0.w + w1 * y1.w;
  *(float4*)(out + (size_t)n * D_MODEL + c) = o;
}

// ---------------------------------------------------------------------------
// Aux loss: deterministic tree reduction of softmax probs over tokens.
// ---------------------------------------------------------------------------
__global__ __launch_bounds__(256)
void moe_aux(const float* __restrict__ probs, float* __restrict__ out) {
  __shared__ float sb[256];
  __shared__ float tot[N_EXP];
  float p[N_EXP];
#pragma unroll
  for (int e = 0; e < N_EXP; ++e) p[e] = 0.0f;
  for (int n = threadIdx.x; n < N_TOK; n += 256)
#pragma unroll
    for (int e = 0; e < N_EXP; ++e) p[e] += probs[(size_t)n * N_EXP + e];
#pragma unroll
  for (int e = 0; e < N_EXP; ++e) {
    sb[threadIdx.x] = p[e];
    __syncthreads();
    for (int o = 128; o > 0; o >>= 1) {
      if (threadIdx.x < o) sb[threadIdx.x] += sb[threadIdx.x + o];
      __syncthreads();
    }
    if (threadIdx.x == 0) tot[e] = sb[0];
    __syncthreads();
  }
  if (threadIdx.x == 0) {
    float L = 0.0f;
#pragma unroll
    for (int e = 0; e < N_EXP; ++e) {
      const float m = tot[e] * (1.0f / N_TOK) - 1.0f / N_EXP;
      L += m * m;
    }
    out[(size_t)N_TOK * D_MODEL] = L;
  }
}

// ---------------------------------------------------------------------------
// Launch
// ---------------------------------------------------------------------------
extern "C" void kernel_launch(void* const* d_in, const int* in_sizes, int n_in,
                              void* d_out, int out_size, void* d_ws, size_t ws_size,
                              hipStream_t stream) {
  const float* x     = (const float*)d_in[0];
  const float* gamma = (const float*)d_in[1];
  const float* beta  = (const float*)d_in[2];
  const float* rw    = (const float*)d_in[3];
  const float* rb    = (const float*)d_in[4];
  const float* w1    = (const float*)d_in[5];
  const float* b1    = (const float*)d_in[6];
  const float* w2    = (const float*)d_in[7];
  const float* b2    = (const float*)d_in[8];
  float* out = (float*)d_out;

  unsigned char* ws = (unsigned char*)d_ws;
  size_t o = 0;
  _Float16* xn  = (_Float16*)(ws + o); o += (size_t)N_TOK * D_MODEL * 2;    // 16 MB
  _Float16* H   = (_Float16*)(ws + o); o += (size_t)N_SLOTS * D_FF * 2;     // 128 MB
  float* Y      = (float*)(ws + o);    o += (size_t)N_SLOTS * D_MODEL * 4;  // 64 MB
  _Float16* w1t = (_Float16*)(ws + o); o += (size_t)N_EXP * D_MODEL * D_FF * 2;  // 64 MB
  _Float16* w2t = (_Float16*)(ws + o); o += (size_t)N_EXP * D_MODEL * D_FF * 2;  // 64 MB
  int* tok_of_slot = (int*)(ws + o);   o += (size_t)N_SLOTS * 4;
  int* tok2slot    = (int*)(ws + o);   o += (size_t)N_SLOTS * 4;
  int* topi        = (int*)(ws + o);   o += (size_t)N_SLOTS * 4;
  float* topw      = (float*)(ws + o); o += (size_t)N_SLOTS * 4;
  float* probs     = (float*)(ws + o); o += (size_t)N_TOK * N_EXP * 4;
  int* meta = (int*)(ws + o);
  int* counts  = meta;
  int* offsets = meta + 8;
  int* cursors = meta + 17;

  // Weight prep (independent of routing): w1[E][D][F] -> w1t[E][F][D] f16,
  // w2[E][F][D] -> w2t[E][D][F] f16.
  moe_prep_w<<<dim3(D_FF / 32, D_MODEL / 32, N_EXP), 256, 0, stream>>>(
      w1, w1t, D_MODEL, D_FF);
  moe_prep_w<<<dim3(D_MODEL / 32, D_FF / 32, N_EXP), 256, 0, stream>>>(
      w2, w2t, D_FF, D_MODEL);

  moe_init<<<1, 32, 0, stream>>>(counts);
  moe_ln_router<<<N_TOK, 256, 0, stream>>>(x, gamma, beta, rw, rb, xn,
                                           counts, topi, topw, probs);
  moe_offsets<<<1, 1, 0, stream>>>(counts, offsets, cursors);
  moe_scatter<<<N_TOK / 256, 256, 0, stream>>>(topi, cursors, tok_of_slot, tok2slot);
  moe_gemm1<<<dim3(D_FF / TILE_N, N_EXP * MAX_TILES), 256, 0, stream>>>(
      xn, w1t, b1, H, counts, offsets, tok_of_slot);
  moe_gemm2<<<dim3(D_MODEL / TILE_N, N_EXP * MAX_TILES), 256, 0, stream>>>(
      H, w2t, b2, Y, counts, offsets);
  moe_combine<<<N_TOK, 256, 0, stream>>>(x, Y, tok2slot, topw, out);
  moe_aux<<<1, 256, 0, stream>>>(probs, out);
}
